// MultiHeadAttention_73272142069863
// MI455X (gfx1250) — matrix operand order repaired and checked by reading, not compiled, not run on hardware
//
#include <hip/hip_runtime.h>

// ---------------------------------------------------------------------------
// MultiHeadAttention (softmax over HEADS) for MI455X / gfx1250, wave32 WMMA.
// Pipeline:
//   cvt inputs f32->bf16, cvt+transpose weights ->
//   bf16 WMMA GEMMs with double-buffered async global->LDS tile staging ->
//   fused scores / softmax-over-heads / attn@V ->
//   output GEMM (f32 out).
// CDNA5 paths exercised: v_wmma_f32_16x16x32_bf16, global_load_async_to_lds,
// s_wait_asynccnt, ds_load_b128 fragment feeds, global_load_b128.
// ---------------------------------------------------------------------------

typedef __attribute__((ext_vector_type(16))) __bf16        v16bf;
typedef __attribute__((ext_vector_type(8)))  float         v8f;
typedef __attribute__((ext_vector_type(4)))  unsigned int  u32x4;

#define DEV __device__ __forceinline__

// f32 -> bf16 bits, round-to-nearest-even
DEV unsigned short f2bf(float f) {
  unsigned u = __builtin_bit_cast(unsigned, f);
  unsigned r = u + 0x7FFFu + ((u >> 16) & 1u);
  return (unsigned short)(r >> 16);
}

union FragU { u32x4 u[2]; v16bf v; };

// Build a 16x(bf16) WMMA fragment from two 16-byte (8-element) runs.
DEV v16bf load_frag(const unsigned short* p0, const unsigned short* p1) {
  FragU f;
  f.u[0] = *(const u32x4*)p0;
  f.u[1] = *(const u32x4*)p1;
  return f.v;
}

DEV v8f wmma_bf16(v16bf a, v16bf b, v8f c) {
  // (neg_a, A, neg_b, B, c_mod, C, reuse_a, reuse_b)
  return __builtin_amdgcn_wmma_f32_16x16x32_bf16(false, a, false, b,
                                                 (short)0, c, false, false);
}

// Async DMA: 16 bytes global -> LDS, tracked by ASYNCcnt (no VGPR data path).
DEV void async_copy_b128(const unsigned short* lds_ptr,
                         const unsigned short* gptr) {
  unsigned loff = (unsigned)(unsigned long long)lds_ptr;  // addrspace(3) off
  unsigned long long ga = (unsigned long long)gptr;
  asm volatile("global_load_async_to_lds_b128 %0, %1, off"
               :: "v"(loff), "v"(ga) : "memory");
}

DEV void wait_async0() {
  asm volatile("s_wait_asynccnt 0x0" ::: "memory");
}

// ---------------------------------------------------------------------------
// Stage 0a: fp32 -> bf16 elementwise (inputs)
// ---------------------------------------------------------------------------
__global__ void cvt_f32_bf16(const float* __restrict__ src,
                             unsigned short* __restrict__ dst, int n) {
  int i = blockIdx.x * blockDim.x + threadIdx.x;
  for (; i < n; i += gridDim.x * blockDim.x) dst[i] = f2bf(src[i]);
}

// ---------------------------------------------------------------------------
// Stage 0b: fp32 [K,N] -> bf16 transposed [N,K] (weights), LDS-tiled 32x32
// ---------------------------------------------------------------------------
__global__ __launch_bounds__(256) void cvt_transpose_f32_bf16(
    const float* __restrict__ src, unsigned short* __restrict__ dst,
    int K, int N) {
  __shared__ float t[32][33];
  const int kb = blockIdx.y * 32, nb = blockIdx.x * 32;
  const int tx = threadIdx.x & 31, ty = threadIdx.x >> 5;  // 8 rows of 32
#pragma unroll
  for (int i = ty; i < 32; i += 8) t[i][tx] = src[(size_t)(kb + i) * N + nb + tx];
  __syncthreads();
#pragma unroll
  for (int i = ty; i < 32; i += 8)
    dst[(size_t)(nb + i) * K + kb + tx] = f2bf(t[tx][i]);
}

// ---------------------------------------------------------------------------
// GEMM  C[M,N] = A[M,K] @ Wt[N,K]^T + bias   (A, Wt bf16; acc f32)
//   OUT_F32:   write fp32 (final projection into d_out), else bf16
//   TRANS_OUT: write bf16 transposed per batch as [B][N][Sdim] (for V)
// Block tile 128x64, BK=32, 256 thr = 8 waves, wave = 32x32 (4 WMMA).
// Both LDS tiles are filled with double-buffered async global->LDS copies.
// ---------------------------------------------------------------------------
template <bool OUT_F32, bool TRANS_OUT>
__global__ __launch_bounds__(256) void gemm_bias(
    const unsigned short* __restrict__ A,    // [M,K] bf16
    const unsigned short* __restrict__ Wt,   // [N,K] bf16 (pre-transposed)
    const float* __restrict__ bias, void* __restrict__ Cp,
    int M, int N, int K, int Sdim) {
  constexpr int BM = 128, BN = 64, BK = 32, PAD = 8;
  __shared__ unsigned short sA[2][BM][BK + PAD];  // [m][k]
  __shared__ unsigned short sB[2][BN][BK + PAD];  // [n][k]

  const int tid = threadIdx.x;
  const int w = tid >> 5, L = tid & 31, lm = L & 15, lh = L >> 4;
  const int m0 = blockIdx.y * BM, n0 = blockIdx.x * BN;
  const int mBase = (w & 3) * 32, nBase = (w >> 2) * 32;
  const int NK = K / BK;

  // Per-thread copy slots: A tile = 512 x 16B chunks (2/thr), W tile = 256 (1/thr)
  const int aRow0 = tid >> 2,         aC0 = (tid & 3) * 8;
  const int aRow1 = (tid + 256) >> 2, aC1 = aC0;
  const int bRow  = tid >> 2,         bC  = (tid & 3) * 8;

  auto issue_tile = [&](int buf, int k0) {
    async_copy_b128(&sA[buf][aRow0][aC0], A + (size_t)(m0 + aRow0) * K + k0 + aC0);
    async_copy_b128(&sA[buf][aRow1][aC1], A + (size_t)(m0 + aRow1) * K + k0 + aC1);
    async_copy_b128(&sB[buf][bRow][bC],  Wt + (size_t)(n0 + bRow) * K + k0 + bC);
  };

  v8f acc[2][2] = {};

  issue_tile(0, 0);
  for (int kt = 0; kt < NK; ++kt) {
    const int c = kt & 1;
    wait_async0();        // own async copies into buf c complete
    __syncthreads();      // everyone's copies complete; prev compute done
    if (kt + 1 < NK) issue_tile(c ^ 1, (kt + 1) * BK);  // overlap with WMMAs

    v16bf af[2], bf[2];
#pragma unroll
    for (int ms = 0; ms < 2; ++ms) {
      const unsigned short* p = &sA[c][mBase + ms * 16 + lm][lh * 8];
      af[ms] = load_frag(p, p + 16);  // A frag: K runs lh*8 and 16+lh*8
    }
#pragma unroll
    for (int ns = 0; ns < 2; ++ns) {
      const unsigned short* p = &sB[c][nBase + ns * 16 + lm][lh * 8];
      bf[ns] = load_frag(p, p + 16);  // B frag: K runs lh*8 and 16+lh*8
    }
#pragma unroll
    for (int ms = 0; ms < 2; ++ms)
#pragma unroll
      for (int ns = 0; ns < 2; ++ns)
        acc[ms][ns] = wmma_bf16(af[ms], bf[ns], acc[ms][ns]);
  }

  // Epilogue: C layout M = r + 8*(lane/16), N = lane%16
#pragma unroll
  for (int ms = 0; ms < 2; ++ms)
#pragma unroll
    for (int ns = 0; ns < 2; ++ns)
#pragma unroll
      for (int r = 0; r < 8; ++r) {
        int row = m0 + mBase + ms * 16 + r + 8 * lh;
        int col = n0 + nBase + ns * 16 + lm;
        float v = acc[ms][ns][r] + bias[col];
        if constexpr (OUT_F32) {
          ((float*)Cp)[(size_t)row * N + col] = v;
        } else if constexpr (TRANS_OUT) {
          ((unsigned short*)Cp)[(size_t)(row / Sdim) * (size_t)N * Sdim +
                                (size_t)col * Sdim + (row % Sdim)] = f2bf(v);
        } else {
          ((unsigned short*)Cp)[(size_t)row * N + col] = f2bf(v);
        }
      }
}

// ---------------------------------------------------------------------------
// Fused scores -> softmax-over-HEADS -> attn @ V
// Block: 256 thr (8 waves) owns one (b, 16-row i-tile); wave w owns heads
// 2w, 2w+1. Softmax over the 16 heads is purely local per (i,j): no online
// recurrence over j is needed, so scores live only in LDS (never HBM).
// ---------------------------------------------------------------------------
__global__ __launch_bounds__(256) void attn_kernel(
    const unsigned short* __restrict__ Qb,   // [B*S, D] bf16
    const unsigned short* __restrict__ Kb,   // [B*S, D] bf16
    const unsigned short* __restrict__ Vt,   // [B, D, S] bf16 (transposed)
    const int* __restrict__ mask,            // [S, S]
    unsigned short* __restrict__ Xb) {       // [B*S, D] bf16
  constexpr int S = 1024, D = 1024, DH = 64;
  constexpr float SCALE = 0.125f;  // 1/sqrt(64)

  __shared__ float          sS[16][32][16];  // [i][j][h]  32 KB
  __shared__ unsigned short sP[16][16][40];  // [h][i][j+pad] 20 KB

  const int tid = threadIdx.x;
  const int w = tid >> 5, L = tid & 31, lm = L & 15, lh = L >> 4;
  const int i0 = blockIdx.x * 16, b = blockIdx.y;

  // Q fragments for this wave's two heads (row tile is fixed -> hoist)
  v16bf qf[2][2];
#pragma unroll
  for (int hh = 0; hh < 2; ++hh) {
    int h = w * 2 + hh;
    const unsigned short* qrow = Qb + (size_t)(b * S + i0 + lm) * D + h * DH;
#pragma unroll
    for (int kk = 0; kk < 2; ++kk)
      qf[hh][kk] = load_frag(qrow + kk * 32 + lh * 8,
                             qrow + kk * 32 + 16 + lh * 8);
  }

  v8f acc[2][4] = {};  // per head: 16(i) x 64(d) output accumulator

  for (int jt = 0; jt < S / 32; ++jt) {
    const int j0 = jt * 32;
    // ---- scores: Q_i @ K_j^T for 2 heads, 2 j-subtiles, K(=DH)=64 -------
#pragma unroll
    for (int hh = 0; hh < 2; ++hh) {
      int h = w * 2 + hh;
#pragma unroll
      for (int jn = 0; jn < 2; ++jn) {
        const unsigned short* krow =
            Kb + (size_t)(b * S + j0 + jn * 16 + lm) * D + h * DH + lh * 8;
        v8f sc = {};
#pragma unroll
        for (int kk = 0; kk < 2; ++kk) {
          v16bf bfK = load_frag(krow + kk * 32, krow + kk * 32 + 16);
          sc = wmma_bf16(qf[hh][kk], bfK, sc);
        }
#pragma unroll
        for (int r = 0; r < 8; ++r)
          sS[r + 8 * lh][jn * 16 + lm][h] = sc[r];
      }
    }
    __syncthreads();

    // ---- softmax over HEADS (16 contiguous f32 per (i,j)) ---------------
    for (int p = tid; p < 512; p += 256) {
      int i = p >> 5, j = p & 31;
      bool keep = mask[(size_t)(i0 + i) * S + (j0 + j)] != 0;
      float v[16], mx = -__builtin_inff();
#pragma unroll
      for (int h = 0; h < 16; ++h) {
        float x = sS[i][j][h] * SCALE;
        if (!keep) x = -__builtin_inff();
        v[h] = x;
        mx = fmaxf(mx, x);
      }
      float e[16], sum = 0.f;
#pragma unroll
      for (int h = 0; h < 16; ++h) { e[h] = __expf(v[h] - mx); sum += e[h]; }
      float inv = 1.0f / sum;
#pragma unroll
      for (int h = 0; h < 16; ++h) sP[h][i][j] = f2bf(e[h] * inv);
    }
    __syncthreads();

    // ---- attn[16x32] @ V[32x64] per head, accumulate over j-tiles -------
#pragma unroll
    for (int hh = 0; hh < 2; ++hh) {
      int h = w * 2 + hh;
      const unsigned short* prow = &sP[h][lm][lh * 8];
      v16bf pf = load_frag(prow, prow + 16);
#pragma unroll
      for (int ds = 0; ds < 4; ++ds) {
        const unsigned short* vrow = Vt + (size_t)b * D * S +
            (size_t)(h * DH + ds * 16 + lm) * S + j0 + lh * 8;
        v16bf vf = load_frag(vrow, vrow + 16);
        acc[hh][ds] = wmma_bf16(pf, vf, acc[hh][ds]);
      }
    }
  }

  // ---- write X[b, i0+i, h*64+d] as bf16 ---------------------------------
#pragma unroll
  for (int hh = 0; hh < 2; ++hh) {
    int h = w * 2 + hh;
#pragma unroll
    for (int ds = 0; ds < 4; ++ds)
#pragma unroll
      for (int r = 0; r < 8; ++r) {
        int i = r + 8 * lh;
        int d = h * DH + ds * 16 + lm;
        Xb[(size_t)(b * S + i0 + i) * D + d] = f2bf(acc[hh][ds][r]);
      }
  }
}

// ---------------------------------------------------------------------------
// Host launcher
// ---------------------------------------------------------------------------
extern "C" void kernel_launch(void* const* d_in, const int* in_sizes, int n_in,
                              void* d_out, int out_size, void* d_ws,
                              size_t ws_size, hipStream_t stream) {
  (void)in_sizes; (void)n_in; (void)out_size; (void)ws_size;
  constexpr int B = 4, S = 1024, D = 1024;
  constexpr int M = B * S;

  const float* query = (const float*)d_in[0];
  const float* key_  = (const float*)d_in[1];
  const float* value = (const float*)d_in[2];
  const int*   mask  = (const int*)d_in[3];
  const float* Wq = (const float*)d_in[4];  const float* bq = (const float*)d_in[5];
  const float* Wk = (const float*)d_in[6];  const float* bk = (const float*)d_in[7];
  const float* Wv = (const float*)d_in[8];  const float* bv = (const float*)d_in[9];
  const float* Wo = (const float*)d_in[10]; const float* bo = (const float*)d_in[11];

  // Workspace layout (64 MB total):
  char* ws = (char*)d_ws;
  const size_t MB = 1u << 20;
  unsigned short* Qb  = (unsigned short*)(ws + 0 * MB);   // 8 MB  [B*S,D] proj Q
  unsigned short* Kb  = (unsigned short*)(ws + 8 * MB);   // 8 MB  [B*S,D] proj K
  unsigned short* Vt  = (unsigned short*)(ws + 16 * MB);  // 8 MB  [B,D,S] proj V^T
  unsigned short* Xb  = (unsigned short*)(ws + 24 * MB);  // 8 MB  [B*S,D] attn out
  unsigned short* Wqb = (unsigned short*)(ws + 32 * MB);  // 2 MB  [N,K] each
  unsigned short* Wkb = (unsigned short*)(ws + 34 * MB);
  unsigned short* Wvb = (unsigned short*)(ws + 36 * MB);
  unsigned short* Wob = (unsigned short*)(ws + 38 * MB);
  unsigned short* Qx  = (unsigned short*)(ws + 40 * MB);  // 8 MB  bf16 inputs
  unsigned short* Kx  = (unsigned short*)(ws + 48 * MB);
  unsigned short* Vx  = (unsigned short*)(ws + 56 * MB);

  // inputs f32 -> bf16 (read once instead of 16x as f32 inside the GEMM)
  cvt_f32_bf16<<<2048, 256, 0, stream>>>(query, Qx, M * D);
  cvt_f32_bf16<<<2048, 256, 0, stream>>>(key_,  Kx, M * D);
  cvt_f32_bf16<<<2048, 256, 0, stream>>>(value, Vx, M * D);

  // weights f32 [K,N] -> bf16 transposed [N,K] so GEMM B tiles are pure copies
  dim3 tg(D / 32, D / 32);
  cvt_transpose_f32_bf16<<<tg, 256, 0, stream>>>(Wq, Wqb, D, D);
  cvt_transpose_f32_bf16<<<tg, 256, 0, stream>>>(Wk, Wkb, D, D);
  cvt_transpose_f32_bf16<<<tg, 256, 0, stream>>>(Wv, Wvb, D, D);
  cvt_transpose_f32_bf16<<<tg, 256, 0, stream>>>(Wo, Wob, D, D);

  dim3 gg(D / 64, M / 128);  // (N tiles, M tiles) = (16, 32)
  gemm_bias<false, false><<<gg, 256, 0, stream>>>(Qx, Wqb, bq, Qb, M, D, D, S);
  gemm_bias<false, false><<<gg, 256, 0, stream>>>(Kx, Wkb, bk, Kb, M, D, D, S);
  gemm_bias<false, true ><<<gg, 256, 0, stream>>>(Vx, Wvb, bv, Vt, M, D, D, S);

  attn_kernel<<<dim3(S / 16, B), 256, 0, stream>>>(Qb, Kb, Vt, mask, Xb);

  gemm_bias<true,  false><<<gg, 256, 0, stream>>>(Xb, Wob, bo, (float*)d_out,
                                                  M, D, D, S);
}
